// VQVAE_54949811585513
// MI455X (gfx1250) — matrix operand order repaired and checked by reading, not compiled, not run on hardware
//
#include <hip/hip_runtime.h>
#include <hip/hip_bf16.h>

typedef _Float16 half_t;
typedef __attribute__((ext_vector_type(16))) _Float16 v16h;
typedef __attribute__((ext_vector_type(8)))  float    v8f;
typedef __attribute__((ext_vector_type(2)))  _Float16 h2v;

#define EPSBN 1e-5f

// ---------------------------------------------------------------------------
__global__ void zero_buf(float* p, int n) {
    int i = blockIdx.x * blockDim.x + threadIdx.x;
    if (i < n) p[i] = 0.0f;
}

// ---------------------------------------------------------------------------
// VALU direct conv (used only for conv1: Cin=1, and convT3: Cout=1 — both
// memory-bound, ~5 FLOP/byte). Bias omitted: cancels in batch-stat BN.
// ---------------------------------------------------------------------------
__device__ __forceinline__ float conv_point(const float* __restrict__ in,
                                            const float* __restrict__ w,
                                            int n, int oc, int oy, int ox,
                                            int Cin, int Hin, int Win) {
    float acc = 0.0f;
    int iy0 = oy * 2 - 1, ix0 = ox * 2 - 1;
    for (int ic = 0; ic < Cin; ++ic) {
        const float* ip = in + ((size_t)n * Cin + ic) * Hin * Win;
        const float* wp = w + ((size_t)oc * Cin + ic) * 16;
#pragma unroll
        for (int ky = 0; ky < 4; ++ky) {
            int iy = iy0 + ky;
            if ((unsigned)iy >= (unsigned)Hin) continue;
#pragma unroll
            for (int kx = 0; kx < 4; ++kx) {
                int ix = ix0 + kx;
                if ((unsigned)ix >= (unsigned)Win) continue;
                acc = fmaf(ip[(size_t)iy * Win + ix], wp[ky * 4 + kx], acc);
            }
        }
    }
    return acc;
}

__device__ __forceinline__ float convT_point(const float* __restrict__ in,
                                             const float* __restrict__ w,
                                             int n, int oc, int oy, int ox,
                                             int Cin, int Hin, int Win, int Cout) {
    float acc = 0.0f;
#pragma unroll
    for (int ky = 0; ky < 4; ++ky) {
        int ty = oy + 1 - ky;
        if (ty < 0 || (ty & 1)) continue;
        int iy = ty >> 1;
        if (iy >= Hin) continue;
#pragma unroll
        for (int kx = 0; kx < 4; ++kx) {
            int tx = ox + 1 - kx;
            if (tx < 0 || (tx & 1)) continue;
            int ix = tx >> 1;
            if (ix >= Win) continue;
            for (int ic = 0; ic < Cin; ++ic) {
                acc = fmaf(in[((size_t)n * Cin + ic) * Hin * Win + (size_t)iy * Win + ix],
                           w[(((size_t)ic * Cout + oc) * 4 + ky) * 4 + kx], acc);
            }
        }
    }
    return acc;
}

template <int TRANSPOSED>
__global__ __launch_bounds__(256) void conv_stats(const float* __restrict__ in,
                                                  const float* __restrict__ w,
                                                  float* __restrict__ ssum,
                                                  float* __restrict__ ssq,
                                                  int N, int Cin, int Hin, int Win,
                                                  int Hout, int Wout, int Cout) {
    int oc = blockIdx.y;
    long total = (long)N * Hout * Wout;
    float ls = 0.0f, lq = 0.0f;
    for (long i = (long)blockIdx.x * blockDim.x + threadIdx.x; i < total;
         i += (long)gridDim.x * blockDim.x) {
        int ox = (int)(i % Wout);
        long t = i / Wout;
        int oy = (int)(t % Hout);
        int n  = (int)(t / Hout);
        float acc = TRANSPOSED ? convT_point(in, w, n, oc, oy, ox, Cin, Hin, Win, Cout)
                               : conv_point(in, w, n, oc, oy, ox, Cin, Hin, Win);
        ls += acc;
        lq += acc * acc;
    }
    __shared__ float rs[256], rq[256];
    rs[threadIdx.x] = ls; rq[threadIdx.x] = lq;
    __syncthreads();
    for (int s = 128; s > 0; s >>= 1) {
        if (threadIdx.x < s) {
            rs[threadIdx.x] += rs[threadIdx.x + s];
            rq[threadIdx.x] += rq[threadIdx.x + s];
        }
        __syncthreads();
    }
    if (threadIdx.x == 0) {
        atomicAdd(&ssum[oc], rs[0]);
        atomicAdd(&ssq[oc], rq[0]);
    }
}

template <int TRANSPOSED>
__global__ __launch_bounds__(256) void conv_apply(const float* __restrict__ in,
                                                  const float* __restrict__ w,
                                                  const float* __restrict__ ssum,
                                                  const float* __restrict__ ssq,
                                                  const float* __restrict__ g,
                                                  const float* __restrict__ be,
                                                  float* __restrict__ out, int relu,
                                                  int N, int Cin, int Hin, int Win,
                                                  int Hout, int Wout, int Cout) {
    long total = (long)N * Cout * Hout * Wout;
    float cnt = (float)((long)N * Hout * Wout);
    for (long i = (long)blockIdx.x * blockDim.x + threadIdx.x; i < total;
         i += (long)gridDim.x * blockDim.x) {
        int ox = (int)(i % Wout);
        long t = i / Wout;
        int oy = (int)(t % Hout); t /= Hout;
        int oc = (int)(t % Cout);
        int n  = (int)(t / Cout);
        float acc = TRANSPOSED ? convT_point(in, w, n, oc, oy, ox, Cin, Hin, Win, Cout)
                               : conv_point(in, w, n, oc, oy, ox, Cin, Hin, Win);
        float mean = ssum[oc] / cnt;
        float var  = ssq[oc] / cnt - mean * mean;
        float v = g[oc] * ((acc - mean) * rsqrtf(var + EPSBN)) + be[oc];
        if (relu) v = fmaxf(v, 0.0f);
        out[i] = v;
    }
}

// ---------------------------------------------------------------------------
// Weight pre-pack for strided conv (Cin=16, k=4): B-fragment order
// wB[((g*8 + s)*32 + lane)*8 + r] = half2 of W at K rows kg+2r(+1), col oc.
// step s: ky = s>>1, kx = (s&1)*2 + tap; k_local = tap*16 + ic.
// ---------------------------------------------------------------------------
__global__ void prep_wconv(const float* __restrict__ w, h2v* __restrict__ wB, int G) {
    int idx = blockIdx.x * blockDim.x + threadIdx.x;
    if (idx >= G * 8 * 32 * 8) return;
    int r = idx & 7;
    int lane = (idx >> 3) & 31;
    int s = (idx >> 8) & 7;
    int g = idx >> 11;
    int n = (lane & 15) + g * 16;
    int kg = (lane < 16) ? 0 : 16;
    h2v pv;
#pragma unroll
    for (int j = 0; j < 2; ++j) {
        int kl = kg + 2 * r + j;
        int tap = kl >> 4, ic = kl & 15;
        int ky = s >> 1, kx = ((s & 1) << 1) + tap;
        half_t hv = (half_t)w[(((size_t)n * 16 + ic) * 4 + ky) * 4 + kx];
        if (j == 0) pv.x = hv; else pv.y = hv;
    }
    wB[idx] = pv;
}

// Weight pre-pack for ConvTranspose phases. Per phase (py,px) the valid taps
// form a dense 2x2 kernel: ky = (1-py)+2*ty, kx = (1-px)+2*tx2.
// wB[(((ph)*steps + s)*32 + lane)*8 + r]; k = s*32 + kg + 2r + j; tap = k/CIN.
__global__ void prep_wconvT(const float* __restrict__ w, h2v* __restrict__ wB,
                            int CIN, int Cout, int steps) {
    int idx = blockIdx.x * blockDim.x + threadIdx.x;
    if (idx >= 4 * steps * 32 * 8) return;
    int r = idx & 7;
    int lane = (idx >> 3) & 31;
    int s = (idx >> 8) % steps;
    int ph = (idx >> 8) / steps;
    int py = ph >> 1, px = ph & 1;
    int n = lane & 15;
    int kg = (lane < 16) ? 0 : 16;
    h2v pv;
#pragma unroll
    for (int j = 0; j < 2; ++j) {
        int k = s * 32 + kg + 2 * r + j;
        int tap = k / CIN, ic = k % CIN;
        int ky = (1 - py) + 2 * (tap >> 1);
        int kx = (1 - px) + 2 * (tap & 1);
        half_t hv = (half_t)w[(((size_t)ic * Cout + n) * 4 + ky) * 4 + kx];
        if (j == 0) pv.x = hv; else pv.y = hv;
    }
    wB[idx] = pv;
}

// ---------------------------------------------------------------------------
// WMMA implicit-GEMM strided conv (k4/s2/p1, Cin=16). One wave = 16 output
// pixels of one row (M), N = G*16 output channels, K = 256 (8 x k32 steps).
// MODE 0: per-channel sum/sumsq of conv output.  MODE 1: BN + ReLU + store.
// ---------------------------------------------------------------------------
template <int G, int MODE>
__global__ __launch_bounds__(256) void conv_wmma(
    const float* __restrict__ in, const h2v* __restrict__ wB,
    float* __restrict__ ssum, float* __restrict__ ssq,
    const float* __restrict__ gam, const float* __restrict__ bet,
    float* __restrict__ out,
    int N, int Hin, int Win, int Hout, int Wout) {
    constexpr int Cout = G * 16;
    const int wave = threadIdx.x >> 5;
    const int lane = threadIdx.x & 31;
    const int l16 = lane & 15;
    const bool hi = lane >= 16;
    const int hioff = hi ? 8 : 0;

    __shared__ half_t tile[8][4 * 36 * 16];  // per-wave (y,x,c) f16 patch
    __shared__ float obuf[8][16 * 17];
    __shared__ float sbin[32], qbin[32];

    if (MODE == 0) {
        if (threadIdx.x < 32) { sbin[threadIdx.x] = 0.f; qbin[threadIdx.x] = 0.f; }
        __syncthreads();
    }

    const int tilesPerRow = Wout >> 4;
    int tileIdx = blockIdx.x * 8 + wave;
    int tx = tileIdx % tilesPerRow;
    int t = tileIdx / tilesPerRow;
    int oy = t % Hout;
    int n  = t / Hout;
    int ox0 = tx << 4;
    int xbase = (ox0 << 1) - 1;
    int ybase = (oy << 1) - 1;

    // ---- stage zero-padded 4 x 36 x 16 input patch ----
    half_t* tw = tile[wave];
    for (int i = lane; i < 4 * 16 * 36; i += 32) {
        int xo = i % 36;
        int t2 = i / 36;
        int ic = t2 & 15;
        int ry = t2 >> 4;
        int iy = ybase + ry, ix = xbase + xo;
        float v = 0.f;
        if ((unsigned)iy < (unsigned)Hin && (unsigned)ix < (unsigned)Win)
            v = in[((size_t)(n * 16 + ic) * Hin + iy) * Win + ix];
        tw[(ry * 36 + xo) * 16 + ic] = (half_t)v;
    }
    __builtin_amdgcn_wave_barrier();   // DS in-order per wave; fence compiler

    v8f acc[G];
    v8f zz = {};
#pragma unroll
    for (int gg = 0; gg < G; ++gg) acc[gg] = zz;

#pragma unroll
    for (int s = 0; s < 8; ++s) {
        int ky = s >> 1, kx0 = (s & 1) << 1;
        v16h a;
#pragma unroll
        for (int r = 0; r < 8; ++r) {
            int tap = hioff >> 3;                // kx offset within step
            int icp = r << 1;                    // even channel pair base
            int xo  = (l16 << 1) + kx0 + tap;    // ix - xbase
            h2v pv = *(const h2v*)&tw[(ky * 36 + xo) * 16 + icp];
            a[2 * r] = pv.x; a[2 * r + 1] = pv.y;
        }
#pragma unroll
        for (int gg = 0; gg < G; ++gg) {
            v16h b;
            const h2v* wp = wB + ((size_t)(gg * 8 + s) * 32 + lane) * 8;
#pragma unroll
            for (int r = 0; r < 8; ++r) { h2v pv = wp[r]; b[2*r] = pv.x; b[2*r+1] = pv.y; }
            acc[gg] = __builtin_amdgcn_wmma_f32_16x16x32_f16(false, a, false, b,
                                                             (short)0, acc[gg],
                                                             false, false);
        }
    }

    if (MODE == 0) {
#pragma unroll
        for (int gg = 0; gg < G; ++gg) {
            float ls = 0.f, lq = 0.f;
#pragma unroll
            for (int v = 0; v < 8; ++v) { float cv = acc[gg][v]; ls += cv; lq += cv * cv; }
            ls += __shfl_xor(ls, 16, 32);        // lanes l and l+16 share oc
            lq += __shfl_xor(lq, 16, 32);
            if (!hi) {
                atomicAdd(&sbin[gg * 16 + l16], ls);
                atomicAdd(&qbin[gg * 16 + l16], lq);
            }
        }
        __syncthreads();
        if (threadIdx.x < Cout) {
            atomicAdd(&ssum[threadIdx.x], sbin[threadIdx.x]);
            atomicAdd(&ssq[threadIdx.x], qbin[threadIdx.x]);
        }
    } else {
        float cnt = (float)N * (float)Hout * (float)Wout;
        float* ow = obuf[wave];
#pragma unroll
        for (int gg = 0; gg < G; ++gg) {
            int oc = gg * 16 + l16;
            float mean = ssum[oc] / cnt;
            float var  = ssq[oc] / cnt - mean * mean;
            float sc = gam[oc] * rsqrtf(var + EPSBN);
            float sh = bet[oc] - mean * sc;
#pragma unroll
            for (int v = 0; v < 8; ++v)
                ow[l16 * 17 + v + hioff] = fmaxf(fmaf(acc[gg][v], sc, sh), 0.f);
            __builtin_amdgcn_wave_barrier();
            for (int i = lane; i < 256; i += 32) {
                int oc_i = i >> 4, m_i = i & 15;
                out[((size_t)(n * Cout + gg * 16 + oc_i) * Hout + oy) * Wout + ox0 + m_i] =
                    ow[oc_i * 17 + m_i];
            }
            __builtin_amdgcn_wave_barrier();
        }
    }
}

// ---------------------------------------------------------------------------
// WMMA ConvTranspose (k4/s2/p1) via 4 sub-pixel phases. Per phase the conv is
// dense 2x2-tap: K = CIN*4 (steps = CIN/8 WMMA k32 steps). Cout = 16.
// ---------------------------------------------------------------------------
template <int CIN, int MODE>
__global__ __launch_bounds__(256) void convT_wmma(
    const float* __restrict__ in, const h2v* __restrict__ wB,
    float* __restrict__ ssum, float* __restrict__ ssq,
    const float* __restrict__ gam, const float* __restrict__ bet,
    float* __restrict__ out,
    int N, int Hin, int Win, int Hout, int Wout) {
    constexpr int STEPS = CIN / 8;
    const int wave = threadIdx.x >> 5;
    const int lane = threadIdx.x & 31;
    const int l16 = lane & 15;
    const bool hi = lane >= 16;
    const int hioff = hi ? 8 : 0;

    __shared__ half_t tile[8][2 * 18 * CIN];
    __shared__ float obuf[8][16 * 17];
    __shared__ float sbin[16], qbin[16];

    if (MODE == 0) {
        if (threadIdx.x < 16) { sbin[threadIdx.x] = 0.f; qbin[threadIdx.x] = 0.f; }
        __syncthreads();
    }

    int Hq = Hout >> 1, Wq = Wout >> 1;
    int tilesPerRow = Wq >> 4;
    int tileIdx = blockIdx.x * 8 + wave;
    int tx = tileIdx % tilesPerRow;
    int t = tileIdx / tilesPerRow;
    int oyq = t % Hq; t /= Hq;
    int ph = t & 3;
    int n  = t >> 2;
    int py = ph >> 1, px = ph & 1;
    int o0 = tx << 4;
    int rbase = oyq - 1 + py;
    int xbase = o0 - 1 + px;

    // ---- stage zero-padded 2 x 18 x CIN input patch ----
    half_t* tw = tile[wave];
    for (int i = lane; i < 2 * 18 * CIN; i += 32) {
        int xo = i % 18;
        int t2 = i / 18;
        int ic = t2 % CIN;
        int ry = t2 / CIN;
        int iy = rbase + ry, ix = xbase + xo;
        float v = 0.f;
        if ((unsigned)iy < (unsigned)Hin && (unsigned)ix < (unsigned)Win)
            v = in[((size_t)(n * CIN + ic) * Hin + iy) * Win + ix];
        tw[(ry * 18 + xo) * CIN + ic] = (half_t)v;
    }
    __builtin_amdgcn_wave_barrier();

    v8f acc = {};
    const h2v* wph = wB + (size_t)ph * STEPS * 32 * 8;
#pragma unroll
    for (int s = 0; s < STEPS; ++s) {
        v16h a;
#pragma unroll
        for (int r = 0; r < 8; ++r) {
            int klp = (r << 1) + (hioff << 1);  // A-frag K pair base
            int k = s * 32 + klp;
            int tap = k / CIN, icp = k % CIN;
            int ry = 1 - (tap >> 1);
            int xo = l16 + 1 - (tap & 1);
            h2v pv = *(const h2v*)&tw[(ry * 18 + xo) * CIN + icp];
            a[2 * r] = pv.x; a[2 * r + 1] = pv.y;
        }
        v16h b;
        const h2v* wp = wph + ((size_t)s * 32 + lane) * 8;
#pragma unroll
        for (int r = 0; r < 8; ++r) { h2v pv = wp[r]; b[2*r] = pv.x; b[2*r+1] = pv.y; }
        acc = __builtin_amdgcn_wmma_f32_16x16x32_f16(false, a, false, b,
                                                     (short)0, acc, false, false);
    }

    if (MODE == 0) {
        float ls = 0.f, lq = 0.f;
#pragma unroll
        for (int v = 0; v < 8; ++v) { float cv = acc[v]; ls += cv; lq += cv * cv; }
        ls += __shfl_xor(ls, 16, 32);
        lq += __shfl_xor(lq, 16, 32);
        if (!hi) {
            atomicAdd(&sbin[l16], ls);
            atomicAdd(&qbin[l16], lq);
        }
        __syncthreads();
        if (threadIdx.x < 16) {
            atomicAdd(&ssum[threadIdx.x], sbin[threadIdx.x]);
            atomicAdd(&ssq[threadIdx.x], qbin[threadIdx.x]);
        }
    } else {
        float cnt = (float)N * (float)Hout * (float)Wout;
        int oc = l16;
        float mean = ssum[oc] / cnt;
        float var  = ssq[oc] / cnt - mean * mean;
        float sc = gam[oc] * rsqrtf(var + EPSBN);
        float sh = bet[oc] - mean * sc;
        float* ow = obuf[wave];
#pragma unroll
        for (int v = 0; v < 8; ++v)
            ow[l16 * 17 + v + hioff] = fmaxf(fmaf(acc[v], sc, sh), 0.f);
        __builtin_amdgcn_wave_barrier();
        int oyo = 2 * oyq + py;
        for (int i = lane; i < 256; i += 32) {
            int oc_i = i >> 4, m_i = i & 15;
            out[((size_t)(n * 16 + oc_i) * Hout + oyo) * Wout + 2 * (o0 + m_i) + px] =
                ow[oc_i * 17 + m_i];
        }
    }
}

// ---------------------------------------------------------------------------
// Codebook prep: f32 [512,32] -> f16 rows + |c|^2
// ---------------------------------------------------------------------------
__global__ __launch_bounds__(32) void cbprep(const float* __restrict__ cb,
                                             half_t* __restrict__ cbh,
                                             float* __restrict__ cnorm) {
    int code = blockIdx.x, k = threadIdx.x;
    float v = cb[code * 32 + k];
    cbh[code * 32 + k] = (half_t)v;
    float s = v * v;
#pragma unroll
    for (int o = 16; o > 0; o >>= 1) s += __shfl_xor(s, o, 32);
    if (k == 0) cnorm[code] = s;
}

// ---------------------------------------------------------------------------
// Quantizer: z.C^T via WMMA; argmin via monotonic-key ds_min_u64.
// ---------------------------------------------------------------------------
__global__ __launch_bounds__(32) void quantize_kernel(const float* __restrict__ qin,
                                                      const half_t* __restrict__ cbh,
                                                      const float* __restrict__ cnorm,
                                                      const float* __restrict__ cbf,
                                                      float* __restrict__ qst,
                                                      int* __restrict__ idx_out,
                                                      float* __restrict__ quant_acc) {
    const int lane = threadIdx.x;
    const int l16  = lane & 15;
    const bool hi  = lane >= 16;
    const int m0   = blockIdx.x * 16;

    __shared__ unsigned long long best[16];
    if (lane < 16) best[lane] = 0xFFFFFFFFFFFFFFFFull;
    __syncthreads();

    int m     = m0 + l16;
    int n_img = m >> 10;
    int rem   = m & 1023;
    int y = rem >> 5, x = rem & 31;
    size_t base = (size_t)n_img * 32768 + (size_t)y * 32 + x;

    v16h a;
    float pnorm = 0.0f;
#pragma unroll
    for (int h = 0; h < 16; ++h) {
        int r = h >> 1;
        int k = 2 * r + (hi ? 16 : 0) + (h & 1);
        float zv = qin[base + (size_t)k * 1024];
        a[h] = (half_t)zv;
        pnorm += zv * zv;
    }
    float znorm = pnorm + __shfl_xor(pnorm, 16, 32);
    float rnorm[8];
#pragma unroll
    for (int v = 0; v < 8; ++v) rnorm[v] = __shfl(znorm, v + (hi ? 8 : 0), 32);

    const int kg = hi ? 16 : 0;
    for (int t = 0; t < 32; ++t) {
        int code0 = t * 16;
        v16h b;
        const half_t* crow = cbh + (size_t)(code0 + l16) * 32 + kg;
#pragma unroll
        for (int r = 0; r < 8; ++r) {
            h2v p = *(const h2v*)(crow + 2 * r);
            b[2 * r] = p.x; b[2 * r + 1] = p.y;
        }
        v8f c = {};
        c = __builtin_amdgcn_wmma_f32_16x16x32_f16(false, a, false, b,
                                                   (short)0, c, false, false);
        float cn = cnorm[code0 + l16];
#pragma unroll
        for (int v = 0; v < 8; ++v) {
            float d2 = rnorm[v] - 2.0f * c[v] + cn;
            unsigned ub = __float_as_uint(d2);
            ub = (ub & 0x80000000u) ? ~ub : (ub | 0x80000000u);
            unsigned long long key =
                ((unsigned long long)ub << 32) | (unsigned)(code0 + l16);
            atomicMin(&best[v + (hi ? 8 : 0)], key);
        }
    }
    __syncthreads();

    if (lane < 16) {
        int code = (int)(best[lane] & 0xFFFFFFFFull);
        idx_out[m0 + lane] = code;
        const float* crow = cbf + (size_t)code * 32;
        float s = 0.0f;
        for (int k = 0; k < 32; ++k) {
            float zc = qin[base + (size_t)k * 1024];
            float cc = crow[k];
            float d = cc - zc;
            s += d * d;
            qst[base + (size_t)k * 1024] = cc;
        }
        atomicAdd(quant_acc, s);
    }
}

// ---------------------------------------------------------------------------
__global__ __launch_bounds__(256) void recon_kernel(const float* __restrict__ x,
                                                    const float* __restrict__ out,
                                                    float* __restrict__ acc, long n) {
    float ls = 0.0f;
    for (long i = (long)blockIdx.x * blockDim.x + threadIdx.x; i < n;
         i += (long)gridDim.x * blockDim.x) {
        float d = x[i] - out[i];
        ls += d * d;
    }
    __shared__ float rs[256];
    rs[threadIdx.x] = ls;
    __syncthreads();
    for (int s = 128; s > 0; s >>= 1) {
        if (threadIdx.x < s) rs[threadIdx.x] += rs[threadIdx.x + s];
        __syncthreads();
    }
    if (threadIdx.x == 0) atomicAdd(acc, rs[0]);
}

__global__ void finalize_kernel(const float* qacc, const float* racc, float* loss) {
    *loss = (*qacc) * 1.2f / 2097152.0f + (*racc) / 4194304.0f;
}

// ---------------------------------------------------------------------------
extern "C" void kernel_launch(void* const* d_in, const int* in_sizes, int n_in,
                              void* d_out, int out_size, void* d_ws, size_t ws_size,
                              hipStream_t stream) {
    const float* x   = (const float*)d_in[0];
    const float* ew1 = (const float*)d_in[1];
    const float* ew2 = (const float*)d_in[3];
    const float* ew3 = (const float*)d_in[5];
    const float* eg1 = (const float*)d_in[7];  const float* ebe1 = (const float*)d_in[8];
    const float* eg2 = (const float*)d_in[9];  const float* ebe2 = (const float*)d_in[10];
    const float* eg3 = (const float*)d_in[11]; const float* ebe3 = (const float*)d_in[12];
    const float* cb  = (const float*)d_in[13];
    const float* dw1 = (const float*)d_in[14];
    const float* dw2 = (const float*)d_in[16];
    const float* dw3 = (const float*)d_in[18];
    const float* dg1 = (const float*)d_in[20]; const float* dbe1 = (const float*)d_in[21];
    const float* dg2 = (const float*)d_in[22]; const float* dbe2 = (const float*)d_in[23];
    const float* dg3 = (const float*)d_in[24]; const float* dbe3 = (const float*)d_in[25];

    float*  h1    = (float*)d_ws;              // 16,777,216 f (reused as decoder d2)
    float*  h2    = h1 + 16777216;             //  4,194,304 f (reused as d1)
    float*  qin   = h2 + 4194304;              //  2,097,152 f
    float*  qst   = qin + 2097152;             //  2,097,152 f
    half_t* cbh   = (half_t*)(qst + 2097152);  // 16,384 halves
    float*  cnorm = (float*)(cbh + 16384);     // 512 f
    float*  stats = cnorm + 512;               // 256 f
    h2v*    wB2   = (h2v*)(stats + 256);       // 2048
    h2v*    wB3   = wB2 + 2048;                // 4096
    h2v*    wBT1  = wB3 + 4096;                // 4096
    h2v*    wBT2  = wBT1 + 4096;               // 2048

    float* out     = (float*)d_out;
    float* lossp   = out + 4194304;
    int*   idx_out = (int*)d_out + 4194305;

    zero_buf<<<1, 256, 0, stream>>>(stats, 208);
    prep_wconv<<<8, 256, 0, stream>>>(ew2, wB2, 1);
    prep_wconv<<<16, 256, 0, stream>>>(ew3, wB3, 2);
    prep_wconvT<<<16, 256, 0, stream>>>(dw1, wBT1, 32, 16, 4);
    prep_wconvT<<<8, 256, 0, stream>>>(dw2, wBT2, 16, 16, 2);
    cbprep<<<512, 32, 0, stream>>>(cb, cbh, cnorm);

    // ---- Encoder ----
    conv_stats<0><<<dim3(512, 16), 256, 0, stream>>>(x, ew1, stats + 0, stats + 16,
                                                     64, 1, 256, 256, 128, 128, 16);
    conv_apply<0><<<8192, 256, 0, stream>>>(x, ew1, stats + 0, stats + 16, eg1, ebe1,
                                            h1, 1, 64, 1, 256, 256, 128, 128, 16);
    conv_wmma<1, 0><<<2048, 256, 0, stream>>>(h1, wB2, stats + 32, stats + 48,
                                              nullptr, nullptr, nullptr,
                                              64, 128, 128, 64, 64);
    conv_wmma<1, 1><<<2048, 256, 0, stream>>>(h1, wB2, stats + 32, stats + 48,
                                              eg2, ebe2, h2, 64, 128, 128, 64, 64);
    conv_wmma<2, 0><<<512, 256, 0, stream>>>(h2, wB3, stats + 64, stats + 96,
                                             nullptr, nullptr, nullptr,
                                             64, 64, 64, 32, 32);
    conv_wmma<2, 1><<<512, 256, 0, stream>>>(h2, wB3, stats + 64, stats + 96,
                                             eg3, ebe3, qin, 64, 64, 64, 32, 32);

    // ---- Quantizer ----
    quantize_kernel<<<4096, 32, 0, stream>>>(qin, cbh, cnorm, cb, qst, idx_out,
                                             stats + 200);

    // ---- Decoder ----
    convT_wmma<32, 0><<<2048, 256, 0, stream>>>(qst, wBT1, stats + 128, stats + 144,
                                                nullptr, nullptr, nullptr,
                                                64, 32, 32, 64, 64);
    convT_wmma<32, 1><<<2048, 256, 0, stream>>>(qst, wBT1, stats + 128, stats + 144,
                                                dg1, dbe1, h2, 64, 32, 32, 64, 64);
    convT_wmma<16, 0><<<8192, 256, 0, stream>>>(h2, wBT2, stats + 160, stats + 176,
                                                nullptr, nullptr, nullptr,
                                                64, 64, 64, 128, 128);
    convT_wmma<16, 1><<<8192, 256, 0, stream>>>(h2, wBT2, stats + 160, stats + 176,
                                                dg2, dbe2, h1, 64, 64, 64, 128, 128);
    conv_stats<1><<<dim3(512, 1), 256, 0, stream>>>(h1, dw3, stats + 192, stats + 193,
                                                    64, 16, 128, 128, 256, 256, 1);
    conv_apply<1><<<8192, 256, 0, stream>>>(h1, dw3, stats + 192, stats + 193, dg3, dbe3,
                                            out, 0, 64, 16, 128, 128, 256, 256, 1);

    // ---- Losses ----
    recon_kernel<<<2048, 256, 0, stream>>>(x, out, stats + 201, 4194304L);
    finalize_kernel<<<1, 1, 0, stream>>>(stats + 200, stats + 201, lossp);
}
